// Attention_3874060501756
// MI455X (gfx1250) — matrix operand order repaired and checked by reading, not compiled, not run on hardware
//
#include <hip/hip_runtime.h>
#include <hip/hip_bf16.h>
#include <math.h>

typedef __attribute__((ext_vector_type(16))) _Float16 v16h;
typedef __attribute__((ext_vector_type(8)))  float    v8f;

#define B_DIM 8
#define C_DIM 684
#define H_DIM 64
#define W_DIM 256
#define HID_DIM 256
#define A_DIM 512
#define KK 11
#define KP 121          // real K of patch GEMM
#define KPAD 128        // padded K
#define HW (H_DIM * W_DIM)          // 16384
#define NPIX (B_DIM * HW)           // 131072

#define PIX_PER_BLK 32
#define WIN_STRIDE 48               // 11x11 window over 32 pixels: 42 cols used
#define WIN_ROWS 12                 // row 11 is an all-zero pad row

// ---------------------------------------------------------------- query
__global__ __launch_bounds__(512) void query_kernel(
    const float* __restrict__ hidden, const float* __restrict__ W_h,
    const float* __restrict__ b_h, float* __restrict__ query) {
  const int a = threadIdx.x;            // 0..511
  const int b = blockIdx.x;             // 0..7
  float s = b_h[a];
  for (int k = 0; k < HID_DIM; ++k)
    s += hidden[b * HID_DIM + k] * W_h[k * A_DIM + a];
  query[b * A_DIM + a] = s;
}

// ------------------------------------------------- M[p,a] = conv_w^T W_att
// stored transposed & padded: Mmat[a][k] (_Float16, k contiguous, 128 per a)
__global__ __launch_bounds__(128) void mmat_kernel(
    const float* __restrict__ conv_w, const float* __restrict__ W_att,
    _Float16* __restrict__ Mmat) {
  const int p = threadIdx.x;            // 0..127
  const int a = blockIdx.x;             // 0..511
  float s = 0.f;
  if (p < KP)
    for (int c = 0; c < A_DIM; ++c)
      s += conv_w[c * KP + p] * W_att[c * A_DIM + a];
  Mmat[a * KPAD + p] = (_Float16)s;
}

// compile-time LDS offset for patch element k (pad rows -> zero row 11)
__device__ constexpr int off_of(int k) {
  return (k < KP) ? ((k / KK) * WIN_STRIDE + (k % KK)) : (11 * WIN_STRIDE);
}

// --------------------------------------------------------- WMMA epilogue
__device__ __forceinline__ void proc_tile(
    const v8f& acc, int a, int b, int h, int wbase, int half,
    const float* __restrict__ query, const float* __restrict__ W_alpha,
    const float* __restrict__ trans, float esum[8]) {
  const float qv = query[b * A_DIM + a];
  const float wv = W_alpha[a];
  const float* tp =
      trans + (((size_t)b * A_DIM + a) * H_DIM + h) * W_DIM + wbase + half * 8;
#pragma unroll
  for (int r = 0; r < 8; ++r)
    esum[r] += tanhf(acc[r] + qv + tp[r]) * wv;
}

// ------------------------------------------------------------- energy core
// Each block: 32 consecutive pixels of one row. 8 waves x 64 a-values each,
// 2 pixel-tiles (A fragments) x 4 a-tiles x 4 K-steps = 32 WMMA per wave.
__global__ __launch_bounds__(256) void energy_kernel(
    const float* __restrict__ alpha_sum, const float* __restrict__ trans,
    const float* __restrict__ query, const _Float16* __restrict__ Mmat,
    const float* __restrict__ W_alpha, const float* __restrict__ b_alpha,
    float* __restrict__ energy_out) {
  __shared__ float win[WIN_ROWS * WIN_STRIDE];
  __shared__ float eacc[PIX_PER_BLK];

  const int tid  = threadIdx.x;
  const int base = blockIdx.x * PIX_PER_BLK;
  const int b    = base >> 14;               // /16384
  const int rem  = base & (HW - 1);
  const int h    = rem >> 8;                 // /256
  const int w0   = rem & 255;

  // stage alpha_sum window (zero-padded; row 11 all zero) into LDS
  for (int idx = tid; idx < WIN_ROWS * WIN_STRIDE; idx += 256) {
    const int row = idx / WIN_STRIDE;
    const int col = idx - row * WIN_STRIDE;
    const int hh = h + row - 5;
    const int ww = w0 + col - 5;
    float v = 0.f;
    if (row < 11 && col < PIX_PER_BLK + KK - 1 &&
        (unsigned)hh < (unsigned)H_DIM && (unsigned)ww < (unsigned)W_DIM)
      v = alpha_sum[b * HW + hh * W_DIM + ww];
    win[idx] = v;
  }
  if (tid < PIX_PER_BLK) eacc[tid] = 0.f;
  __syncthreads();

  const int lane = tid & 31;
  const int wave = tid >> 5;                 // 0..7
  const int half = lane >> 4;                // 0/1
  const int nl   = lane & 15;                // N lane (B/D), M row (A)
  const int aBase = wave * 64;

  v8f acc0[4] = {v8f{}, v8f{}, v8f{}, v8f{}};   // pixel tile 0 (m = 0..15)
  v8f acc1[4] = {v8f{}, v8f{}, v8f{}, v8f{}};   // pixel tile 1 (m = 16..31)

#pragma unroll
  for (int kt = 0; kt < 4; ++kt) {
    // ---- A operands: both candidate offsets are compile-time constants,
    //      select by runtime `half` (one cndmask), pad K handled by zero row.
    v16h a0, a1;
#pragma unroll
    for (int e = 0; e < 16; ++e) {
      const int k0  = kt * 32 + e;
      const int o0  = off_of(k0);      // half == 0
      const int o1  = off_of(k0 + 16); // half == 1
      const int off = (half ? o1 : o0) + nl;
      a0[e] = (_Float16)win[off];
      a1[e] = (_Float16)win[off + 16];
    }
    // ---- B operands: contiguous 32B per lane (k_local = 16*half + i)
    const _Float16* mp = Mmat + kt * 32 + 16 * half;
#pragma unroll
    for (int t = 0; t < 4; ++t) {
      const v16h bv = *(const v16h*)(mp + (size_t)(aBase + t * 16 + nl) * KPAD);
      acc0[t] = __builtin_amdgcn_wmma_f32_16x16x32_f16(false, a0, false, bv,
                                                       (short)0, acc0[t], false, false);
      acc1[t] = __builtin_amdgcn_wmma_f32_16x16x32_f16(false, a1, false, bv,
                                                       (short)0, acc1[t], false, false);
    }
  }

  // ---- epilogue: tanh(query + coverage + trans) . W_alpha
  float esum0[8] = {0.f, 0.f, 0.f, 0.f, 0.f, 0.f, 0.f, 0.f};
  float esum1[8] = {0.f, 0.f, 0.f, 0.f, 0.f, 0.f, 0.f, 0.f};
#pragma unroll
  for (int t = 0; t < 4; ++t) {
    const int a = aBase + t * 16 + nl;
    proc_tile(acc0[t], a, b, h, w0,      half, query, W_alpha, trans, esum0);
    proc_tile(acc1[t], a, b, h, w0 + 16, half, query, W_alpha, trans, esum1);
  }

  // reduce over the 16 N-lanes of each half (pixel = mtile*16 + half*8 + r)
#pragma unroll
  for (int r = 0; r < 8; ++r) {
    float v = esum0[r];
    v += __shfl_xor(v, 1, 32);
    v += __shfl_xor(v, 2, 32);
    v += __shfl_xor(v, 4, 32);
    v += __shfl_xor(v, 8, 32);
    esum0[r] = v;
    float u = esum1[r];
    u += __shfl_xor(u, 1, 32);
    u += __shfl_xor(u, 2, 32);
    u += __shfl_xor(u, 4, 32);
    u += __shfl_xor(u, 8, 32);
    esum1[r] = u;
  }
  if (nl == 0) {
#pragma unroll
    for (int r = 0; r < 8; ++r) {
      atomicAdd(&eacc[half * 8 + r], esum0[r]);
      atomicAdd(&eacc[16 + half * 8 + r], esum1[r]);
    }
  }
  __syncthreads();
  if (tid < PIX_PER_BLK) energy_out[base + tid] = eacc[tid] + b_alpha[0];
}

// ------------------------------------------------------------ reductions
__global__ __launch_bounds__(256) void maxred1(const float* __restrict__ energy,
                                               float* __restrict__ pmax) {
  __shared__ float s[256];
  const int tid = threadIdx.x;
  const int base = blockIdx.x * 1024;
  float m = -3.4e38f;
#pragma unroll
  for (int j = 0; j < 4; ++j) m = fmaxf(m, energy[base + j * 256 + tid]);
  s[tid] = m; __syncthreads();
  for (int off = 128; off > 0; off >>= 1) {
    if (tid < off) s[tid] = fmaxf(s[tid], s[tid + off]);
    __syncthreads();
  }
  if (tid == 0) pmax[blockIdx.x] = s[0];
}

__global__ __launch_bounds__(128) void maxred2(const float* __restrict__ pmax,
                                               float* __restrict__ gmax,
                                               float* __restrict__ denom) {
  __shared__ float s[128];
  const int tid = threadIdx.x;
  s[tid] = pmax[tid]; __syncthreads();
  for (int off = 64; off > 0; off >>= 1) {
    if (tid < off) s[tid] = fmaxf(s[tid], s[tid + off]);
    __syncthreads();
  }
  if (tid == 0) gmax[0] = s[0];
  if (tid < B_DIM) denom[tid] = 0.f;
}

__global__ __launch_bounds__(256) void exp_kernel(
    const float* __restrict__ energy, const float* __restrict__ mask,
    const float* __restrict__ gmax, float* __restrict__ eexp,
    float* __restrict__ denom) {
  __shared__ float s[256];
  const int tid = threadIdx.x;
  const int i = blockIdx.x * 256 + tid;
  const int b = i >> 14;
  const float e = expf(energy[i] - gmax[0]) * mask[i];
  eexp[i] = e;
  s[tid] = e; __syncthreads();
  for (int off = 128; off > 0; off >>= 1) {
    if (tid < off) s[tid] += s[tid + off];
    __syncthreads();
  }
  if (tid == 0) atomicAdd(&denom[b], s[0]);
}

__global__ __launch_bounds__(256) void alpha_kernel(
    const float* __restrict__ eexp, const float* __restrict__ denom,
    const float* __restrict__ alpha_sum, float* __restrict__ out_alpha,
    float* __restrict__ out_nas, float* __restrict__ weight) {
  const int i = blockIdx.x * 256 + threadIdx.x;
  const int b = i >> 14;
  const float al = eexp[i] / (denom[b] + 1e-10f);
  out_alpha[i] = al;
  out_nas[i] = al + alpha_sum[i];
  weight[i] = (al > 0.02f) ? al : 0.f;
}

__global__ __launch_bounds__(256) void ctx_kernel(
    const float* __restrict__ weight, const float* __restrict__ feat,
    float* __restrict__ out_ctx) {
  __shared__ float sm[256];
  const int bc = blockIdx.x;               // b*684 + c
  const int b = bc / C_DIM;
  const float4* wp = (const float4*)(weight + (size_t)b * HW);
  const float4* fp = (const float4*)(feat + (size_t)bc * HW);
  float s = 0.f;
  for (int j = threadIdx.x; j < HW / 4; j += 256) {
    const float4 w = wp[j];
    const float4 f = fp[j];
    s += w.x * f.x + w.y * f.y + w.z * f.z + w.w * f.w;
  }
  sm[threadIdx.x] = s; __syncthreads();
  for (int off = 128; off > 0; off >>= 1) {
    if (threadIdx.x < off) sm[threadIdx.x] += sm[threadIdx.x + off];
    __syncthreads();
  }
  if (threadIdx.x == 0) out_ctx[bc] = sm[0];
}

// ---------------------------------------------------------------- launch
extern "C" void kernel_launch(void* const* d_in, const int* in_sizes, int n_in,
                              void* d_out, int out_size, void* d_ws, size_t ws_size,
                              hipStream_t stream) {
  const float* cnn_features = (const float*)d_in[0];   // [8,684,64,256]
  const float* trans        = (const float*)d_in[1];   // [8,512,64,256]
  const float* hidden       = (const float*)d_in[2];   // [8,256]
  const float* alpha_sum    = (const float*)d_in[3];   // [8,1,64,256]
  const float* image_mask   = (const float*)d_in[4];   // [8,1,64,256]
  const float* W_h          = (const float*)d_in[5];   // [256,512]
  const float* b_h          = (const float*)d_in[6];   // [512]
  const float* conv_w       = (const float*)d_in[7];   // [512,1,11,11]
  const float* W_att        = (const float*)d_in[8];   // [512,512]
  const float* W_alpha      = (const float*)d_in[9];   // [512,1]
  const float* b_alpha      = (const float*)d_in[10];  // [1]

  // outputs: context [8,684] | alpha [8,64,256] | new_alpha_sum [8,1,64,256]
  float* out_ctx   = (float*)d_out;
  float* out_alpha = out_ctx + B_DIM * C_DIM;
  float* out_nas   = out_alpha + NPIX;

  // workspace layout (bytes)
  char* ws = (char*)d_ws;
  float*     query  = (float*)(ws + 0);                        // 4096 f
  _Float16*  Mmat   = (_Float16*)(ws + 16384);                 // 512*128 h
  float*     energy = (float*)(ws + 16384 + 131072);           // 131072 f
  float*     eexp   = energy + NPIX;
  float*     weight = eexp + NPIX;
  float*     pmax   = weight + NPIX;                           // 128 f
  float*     gmax   = pmax + 128;                              // 1 f
  float*     denom  = gmax + 1;                                // 8 f

  query_kernel<<<B_DIM, A_DIM, 0, stream>>>(hidden, W_h, b_h, query);
  mmat_kernel<<<A_DIM, KPAD, 0, stream>>>(conv_w, W_att, Mmat);
  energy_kernel<<<NPIX / PIX_PER_BLK, 256, 0, stream>>>(alpha_sum, trans, query,
                                                        Mmat, W_alpha, b_alpha,
                                                        energy);
  maxred1<<<128, 256, 0, stream>>>(energy, pmax);
  maxred2<<<1, 128, 0, stream>>>(pmax, gmax, denom);
  exp_kernel<<<NPIX / 256, 256, 0, stream>>>(energy, image_mask, gmax, eexp, denom);
  alpha_kernel<<<NPIX / 256, 256, 0, stream>>>(eexp, denom, alpha_sum,
                                               out_alpha, out_nas, weight);
  ctx_kernel<<<B_DIM * C_DIM, 256, 0, stream>>>(weight, cnn_features, out_ctx);
}
